// MambaV7_39556648796760
// MI455X (gfx1250) — matrix operand order repaired and checked by reading, not compiled, not run on hardware
//
#include <hip/hip_runtime.h>
#include <hip/hip_bf16.h>

typedef __attribute__((ext_vector_type(16))) __bf16 v16bf;
typedef __attribute__((ext_vector_type(8)))  float  v8f;

#define LDK 72   // 64 + 8 bf16 pad: 144B row stride -> 16B-aligned b128 async stores

__device__ __forceinline__ unsigned short f2bf(float f) {
  unsigned int u = __float_as_uint(f);
  u += 0x7FFFu + ((u >> 16) & 1u);          // round-to-nearest-even
  return (unsigned short)(u >> 16);
}
__device__ __forceinline__ float bf2f(unsigned short h) {
  return __uint_as_float(((unsigned int)h) << 16);
}

// ---------------------------------------------------------------------------
// Generic NT GEMM: C[M,N] = A[M,K](bf16, row stride lda) * W[N,K](bf16)^T
// optional bias[N], act (1 = softplus), fp32 and/or bf16 output (row stride ldc)
// 256 threads = 8 waves; block tile 128x64; per-wave 32x32 (4 accumulators,
// 8 WMMAs per 64-wide K stage). Full tiles staged with async global->LDS.
// ---------------------------------------------------------------------------
__global__ __launch_bounds__(256)
void k_gemm_bf16(const unsigned short* __restrict__ A, long long lda,
                 const unsigned short* __restrict__ W,
                 const float* __restrict__ bias, int act,
                 float* __restrict__ outF, unsigned short* __restrict__ outB,
                 int M, int N, int K, int ldc)
{
  __shared__ unsigned short sA[128 * LDK];
  __shared__ unsigned short sW[64 * LDK];

  const int tid  = threadIdx.x;
  const int lane = tid & 31;
  const int wid  = tid >> 5;
  const int mw   = wid >> 1;            // 0..3  (32-row slab)
  const int nw   = wid & 1;             // 0..1  (32-col slab)
  const int m16  = lane & 15;
  const int hi   = lane >> 4;

  const int blockRow = blockIdx.y * 128;
  const int blockCol = blockIdx.x * 64;

  v8f acc00 = {}, acc01 = {}, acc10 = {}, acc11 = {};

  const int nch = (K + 63) >> 6;
  for (int ch = 0; ch < nch; ++ch) {
    const int kc = ch << 6;
    __syncthreads();

    const bool full = (blockRow + 128 <= M) && (blockCol + 64 <= N) && (kc + 64 <= K);
    if (full) {
      // -------- async global -> LDS staging (ASYNCcnt-tracked, no VGPR data)
      #pragma unroll
      for (int t = 0; t < 4; ++t) {               // A: 128 rows x 8 groups
        const int task = tid + t * 256;
        const int r = task >> 3, gcol = (task & 7) << 3;
        const unsigned short* gp = A + (size_t)(blockRow + r) * lda + kc + gcol;
        const unsigned loff = (unsigned)(size_t)&sA[r * LDK + gcol];
        asm volatile("global_load_async_to_lds_b128 %0, %1, off"
                     :: "v"(loff), "v"(gp) : "memory");
      }
      #pragma unroll
      for (int t = 0; t < 2; ++t) {               // W: 64 rows x 8 groups
        const int task = tid + t * 256;
        const int r = task >> 3, gcol = (task & 7) << 3;
        const unsigned short* gp = W + (size_t)(blockCol + r) * K + kc + gcol;
        const unsigned loff = (unsigned)(size_t)&sW[r * LDK + gcol];
        asm volatile("global_load_async_to_lds_b128 %0, %1, off"
                     :: "v"(loff), "v"(gp) : "memory");
      }
      asm volatile("s_wait_asynccnt 0x0" ::: "memory");
    } else {
      // -------- guarded fallback staging (M/N/K edges), zero-padded
      #pragma unroll
      for (int t = 0; t < 4; ++t) {
        const int task = tid + t * 256;
        const int rr = task >> 3, gcol = (task & 7) << 3;
        const int gm = blockRow + rr;
        const int gk = kc + gcol;
        unsigned int u0 = 0, u1 = 0, u2 = 0, u3 = 0;
        if (gm < M) {
          if (gk + 8 <= K) {
            const uint4 q = *(const uint4*)(A + (size_t)gm * lda + gk);
            u0 = q.x; u1 = q.y; u2 = q.z; u3 = q.w;
          } else {
            unsigned short tb[8] = {0,0,0,0,0,0,0,0};
            for (int i = 0; i < 8; ++i) if (gk + i < K) tb[i] = A[(size_t)gm * lda + gk + i];
            u0 = tb[0] | ((unsigned)tb[1] << 16); u1 = tb[2] | ((unsigned)tb[3] << 16);
            u2 = tb[4] | ((unsigned)tb[5] << 16); u3 = tb[6] | ((unsigned)tb[7] << 16);
          }
          if (gk + 72 <= K) __builtin_prefetch(A + (size_t)gm * lda + gk + 64, 0, 1);
        }
        uint2* p = (uint2*)&sA[rr * LDK + gcol];
        p[0] = make_uint2(u0, u1); p[1] = make_uint2(u2, u3);
      }
      #pragma unroll
      for (int t = 0; t < 2; ++t) {
        const int task = tid + t * 256;
        const int rr = task >> 3, gcol = (task & 7) << 3;
        const int gn = blockCol + rr;
        const int gk = kc + gcol;
        unsigned int u0 = 0, u1 = 0, u2 = 0, u3 = 0;
        if (gn < N) {
          if (gk + 8 <= K) {
            const uint4 q = *(const uint4*)(W + (size_t)gn * K + gk);
            u0 = q.x; u1 = q.y; u2 = q.z; u3 = q.w;
          } else {
            unsigned short tb[8] = {0,0,0,0,0,0,0,0};
            for (int i = 0; i < 8; ++i) if (gk + i < K) tb[i] = W[(size_t)gn * K + gk + i];
            u0 = tb[0] | ((unsigned)tb[1] << 16); u1 = tb[2] | ((unsigned)tb[3] << 16);
            u2 = tb[4] | ((unsigned)tb[5] << 16); u3 = tb[6] | ((unsigned)tb[7] << 16);
          }
        }
        uint2* p = (uint2*)&sW[rr * LDK + gcol];
        p[0] = make_uint2(u0, u1); p[1] = make_uint2(u2, u3);
      }
    }
    __syncthreads();

    #pragma unroll
    for (int kk = 0; kk < 64; kk += 32) {
      union Frag { v16bf v; unsigned int u[8]; } a0, a1, w0, w1;
      const int ar0 = (mw << 5) + m16;      // wave's first 16-row tile
      const int ar1 = ar0 + 16;             // second 16-row tile
      const int wr0 = (nw << 5) + m16;      // first 16-col tile (W rows = N)
      const int wr1 = wr0 + 16;
      #pragma unroll
      for (int v = 0; v < 8; ++v) {
        // A (16x32 bf16): lanes 0-15 K=0..7/16..23, lanes 16-31 K=8..15/24..31
        const int ka = ((v & 4) ? 16 : 0) + (hi << 3) + ((v & 3) << 1);
        a0.u[v] = *(const unsigned int*)&sA[ar0 * LDK + kk + ka];
        a1.u[v] = *(const unsigned int*)&sA[ar1 * LDK + kk + ka];
        // B (32x16 bf16): lanes 0-15 K=0..7/16..23, lanes 16-31 K=8..15/24..31
        const int kb = ((v & 4) ? 16 : 0) + (hi << 3) + ((v & 3) << 1);
        w0.u[v] = *(const unsigned int*)&sW[wr0 * LDK + kk + kb];
        w1.u[v] = *(const unsigned int*)&sW[wr1 * LDK + kk + kb];
      }
      acc00 = __builtin_amdgcn_wmma_f32_16x16x32_bf16(false, a0.v, false, w0.v, (short)0, acc00, false, false);
      acc01 = __builtin_amdgcn_wmma_f32_16x16x32_bf16(false, a0.v, false, w1.v, (short)0, acc01, false, false);
      acc10 = __builtin_amdgcn_wmma_f32_16x16x32_bf16(false, a1.v, false, w0.v, (short)0, acc10, false, false);
      acc11 = __builtin_amdgcn_wmma_f32_16x16x32_bf16(false, a1.v, false, w1.v, (short)0, acc11, false, false);
    }
  }

  // epilogue: C/D layout -> VGPR r holds row r (lanes<16) / r+8 (lanes>=16), col = lane%16
  #pragma unroll
  for (int i = 0; i < 2; ++i) {
    #pragma unroll
    for (int j = 0; j < 2; ++j) {
      const v8f& acc = i ? (j ? acc11 : acc10) : (j ? acc01 : acc00);
      const int gn = blockCol + (nw << 5) + (j << 4) + m16;
      #pragma unroll
      for (int r = 0; r < 8; ++r) {
        const int gm = blockRow + (mw << 5) + (i << 4) + r + (hi << 3);
        if (gm < M && gn < N) {
          float val = acc[r];
          if (bias) val += bias[gn];
          if (act == 1) val = (val > 20.f) ? val : log1pf(expf(val));
          if (outF) outF[(size_t)gm * ldc + gn] = val;
          if (outB) outB[(size_t)gm * ldc + gn] = f2bf(val);
        }
      }
    }
  }
}

// ---------------------------------------------------------------------------
// Encoder + LayerNorm fused: one block per token, 256 threads x 2 outputs.
// ---------------------------------------------------------------------------
__global__ __launch_bounds__(256)
void k_enc_ln(const float* __restrict__ x, const float* __restrict__ ew,
              const float* __restrict__ eb, const float* __restrict__ g,
              const float* __restrict__ be, unsigned short* __restrict__ out)
{
  const int row = blockIdx.x;
  const int tid = threadIdx.x;
  __shared__ float xs[32];
  __shared__ float red[256];
  if (tid < 32) xs[tid] = x[(size_t)row * 32 + tid];
  __syncthreads();
  float v0 = eb[tid], v1 = eb[tid + 256];
  #pragma unroll
  for (int k = 0; k < 32; ++k) {
    v0 += xs[k] * ew[tid * 32 + k];
    v1 += xs[k] * ew[(tid + 256) * 32 + k];
  }
  v0 = fmaxf(v0, 0.f); v1 = fmaxf(v1, 0.f);
  red[tid] = v0 + v1;
  __syncthreads();
  for (int s = 128; s > 0; s >>= 1) { if (tid < s) red[tid] += red[tid + s]; __syncthreads(); }
  const float mean = red[0] * (1.f / 512.f);
  __syncthreads();
  red[tid] = v0 * v0 + v1 * v1;
  __syncthreads();
  for (int s = 128; s > 0; s >>= 1) { if (tid < s) red[tid] += red[tid + s]; __syncthreads(); }
  const float var = red[0] * (1.f / 512.f) - mean * mean;
  const float rs = rsqrtf(var + 1e-5f);
  out[(size_t)row * 512 + tid]       = f2bf((v0 - mean) * rs * g[tid] + be[tid]);
  out[(size_t)row * 512 + tid + 256] = f2bf((v1 - mean) * rs * g[tid + 256] + be[tid + 256]);
}

// ---------------------------------------------------------------------------
// Causal depthwise conv (K=4) + SiLU; reads xi = xz[:, :, 0:1024] (bf16).
// ---------------------------------------------------------------------------
__global__ __launch_bounds__(256)
void k_conv_silu(const unsigned short* __restrict__ xz, const float* __restrict__ cw,
                 const float* __restrict__ cb, unsigned short* __restrict__ xc, int S)
{
  const size_t idx = (size_t)blockIdx.x * 256 + threadIdx.x;   // over B*S*1024
  const int d = (int)(idx & 1023);
  const size_t bs = idx >> 10;
  const int s = (int)(bs % (size_t)S);
  float acc = cb[d];
  #pragma unroll
  for (int k = 0; k < 4; ++k) {
    const int ss = s - 3 + k;
    if (ss >= 0) acc += bf2f(xz[(bs - (size_t)(3 - k)) * 2048 + d]) * cw[d * 4 + k];
  }
  acc = acc / (1.f + expf(-acc));           // SiLU
  xc[bs * 1024 + d] = f2bf(acc);
}

__global__ void k_slice_dt(const float* __restrict__ proj, unsigned short* __restrict__ dtp, int n)
{
  const int i = blockIdx.x * 256 + threadIdx.x;
  if (i < n) {
    const int j = i & 31;
    const size_t m = (size_t)(i >> 5);
    dtp[i] = f2bf(proj[m * 64 + j]);
  }
}

__global__ void k_cast_bf16(const float* __restrict__ s, unsigned short* __restrict__ d, int n)
{
  const int i = blockIdx.x * 256 + threadIdx.x;
  if (i < n) d[i] = f2bf(s[i]);
}

// ---------------------------------------------------------------------------
// Selective scan: one thread per (batch, channel); 16 states in registers.
// Fuses D-skip connection and SiLU(z) gating; writes gated y as bf16.
// ---------------------------------------------------------------------------
__global__ __launch_bounds__(256)
void k_scan(const unsigned short* __restrict__ dt_bf, const unsigned short* __restrict__ xc,
            const float* __restrict__ proj, const unsigned short* __restrict__ xz,
            const float* __restrict__ A_log, const float* __restrict__ Dp,
            unsigned short* __restrict__ yg, int S)
{
  const int idx = blockIdx.x * 256 + threadIdx.x;   // B*1024
  const int d = idx & 1023;
  const int b = idx >> 10;
  float Av[16], h[16];
  #pragma unroll
  for (int i = 0; i < 16; ++i) { Av[i] = -expf(A_log[d * 16 + i]); h[i] = 0.f; }
  const float Dd = Dp[d];
  const size_t base = (size_t)b * S;
  for (int s = 0; s < S; ++s) {
    const size_t rsi = base + s;
    const float dt = bf2f(dt_bf[rsi * 1024 + d]);
    const float xv = bf2f(xc[rsi * 1024 + d]);
    const float* bc = proj + rsi * 64;              // B at [32..48), C at [48..64)
    const float dtx = dt * xv;
    float y = 0.f;
    #pragma unroll
    for (int i = 0; i < 16; ++i) {
      const float dA = expf(dt * Av[i]);
      h[i] = dA * h[i] + dtx * bc[32 + i];
      y += h[i] * bc[48 + i];
    }
    y += xv * Dd;
    const float z = bf2f(xz[rsi * 2048 + 1024 + d]);
    yg[rsi * 1024 + d] = f2bf(y * (z / (1.f + expf(-z))));
  }
}

// ---------------------------------------------------------------------------
// Final LN (last token) + 2-layer head + skip connection. One block per batch.
// ---------------------------------------------------------------------------
__global__ __launch_bounds__(512)
void k_head(const float* __restrict__ hl, const float* __restrict__ ng, const float* __restrict__ nb,
            const float* __restrict__ w1, const float* __restrict__ b1,
            const float* __restrict__ w2, const float* __restrict__ b2,
            const float* __restrict__ x, const float* __restrict__ sw, const float* __restrict__ sb,
            float* __restrict__ out, int S)
{
  const int b = blockIdx.x;
  const int tid = threadIdx.x;
  __shared__ float sh[512];
  __shared__ float red[512];
  __shared__ float y1[32];
  const float v = hl[(size_t)b * 512 + tid];
  red[tid] = v; __syncthreads();
  for (int s = 256; s > 0; s >>= 1) { if (tid < s) red[tid] += red[tid + s]; __syncthreads(); }
  const float mean = red[0] * (1.f / 512.f);
  __syncthreads();
  red[tid] = v * v; __syncthreads();
  for (int s = 256; s > 0; s >>= 1) { if (tid < s) red[tid] += red[tid + s]; __syncthreads(); }
  const float var = red[0] * (1.f / 512.f) - mean * mean;
  const float rs = rsqrtf(var + 1e-5f);
  sh[tid] = (v - mean) * rs * ng[tid] + nb[tid];
  __syncthreads();
  if (tid < 32) {
    float a = b1[tid];
    for (int k = 0; k < 512; ++k) a += sh[k] * w1[tid * 512 + k];
    y1[tid] = fmaxf(a, 0.f);
  }
  __syncthreads();
  if (tid == 0) {
    float o = b2[0];
    for (int j = 0; j < 32; ++j) o += y1[j] * w2[j];
    float sk = sb[0];
    const float* xr = x + ((size_t)b * S + (S - 1)) * 32;
    for (int k = 0; k < 32; ++k) sk += xr[k] * sw[k];
    out[b] = o + sk;
  }
}

// ---------------------------------------------------------------------------
extern "C" void kernel_launch(void* const* d_in, const int* in_sizes, int n_in,
                              void* d_out, int out_size, void* d_ws, size_t ws_size,
                              hipStream_t stream)
{
  (void)in_sizes; (void)n_in; (void)out_size; (void)ws_size;
  const int B = 32, S = 1024, DI = 1024;
  const int M = B * S;

  const float* x      = (const float*)d_in[0];
  const float* enc_w  = (const float*)d_in[1];
  const float* enc_b  = (const float*)d_in[2];
  const float* ln1_g  = (const float*)d_in[3];
  const float* ln1_b  = (const float*)d_in[4];
  const float* norm_g = (const float*)d_in[23];
  const float* norm_b = (const float*)d_in[24];
  const float* h1_w   = (const float*)d_in[25];
  const float* h1_b   = (const float*)d_in[26];
  const float* h2_w   = (const float*)d_in[27];
  const float* h2_b   = (const float*)d_in[28];
  const float* skip_w = (const float*)d_in[29];
  const float* skip_b = (const float*)d_in[30];

  char* ws = (char*)d_ws;
  size_t off = 0;
  auto alloc = [&](size_t bytes) -> void* {
    off = (off + 255) & ~(size_t)255;
    void* p = ws + off;
    off += bytes;
    return p;
  };

  unsigned short* wb_in[2]  = { (unsigned short*)alloc((size_t)2048*512*2), (unsigned short*)alloc((size_t)2048*512*2) };
  unsigned short* wb_xp[2]  = { (unsigned short*)alloc((size_t)64*1024*2),  (unsigned short*)alloc((size_t)64*1024*2) };
  unsigned short* wb_dt[2]  = { (unsigned short*)alloc((size_t)1024*32*2),  (unsigned short*)alloc((size_t)1024*32*2) };
  unsigned short* wb_out[2] = { (unsigned short*)alloc((size_t)512*1024*2), (unsigned short*)alloc((size_t)512*1024*2) };
  unsigned short* h_bf   = (unsigned short*)alloc((size_t)M * 512 * 2);
  unsigned short* h2_bf  = (unsigned short*)alloc((size_t)M * 512 * 2);
  unsigned short* xz_bf  = (unsigned short*)alloc((size_t)M * 2048 * 2);
  unsigned short* xc_bf  = (unsigned short*)alloc((size_t)M * 1024 * 2);
  float*          proj_f = (float*)         alloc((size_t)M * 64 * 4);
  unsigned short* dtp_bf = (unsigned short*)alloc((size_t)M * 32 * 2);
  unsigned short* dt_bf  = (unsigned short*)alloc((size_t)M * 1024 * 2);
  unsigned short* yg_bf  = (unsigned short*)alloc((size_t)M * 1024 * 2);
  float*          hlast  = (float*)         alloc((size_t)B * 512 * 4);

  auto castw = [&](const float* src, unsigned short* dst, int n) {
    k_cast_bf16<<<(n + 255) / 256, 256, 0, stream>>>(src, dst, n);
  };
  for (int blk = 0; blk < 2; ++blk) {
    const int o = 5 + 9 * blk;
    castw((const float*)d_in[o + 0], wb_in[blk],  2048 * 512);
    castw((const float*)d_in[o + 3], wb_xp[blk],  64 * 1024);
    castw((const float*)d_in[o + 4], wb_dt[blk],  1024 * 32);
    castw((const float*)d_in[o + 8], wb_out[blk], 512 * 1024);
  }

  k_enc_ln<<<M, 256, 0, stream>>>(x, enc_w, enc_b, ln1_g, ln1_b, h_bf);

  auto gemm = [&](const unsigned short* A, long long lda, const unsigned short* W,
                  const float* bias, int act, float* outF, unsigned short* outB,
                  int Mm, int Nn, int Kk, int ldc) {
    dim3 grid((Nn + 63) / 64, (Mm + 127) / 128);
    k_gemm_bf16<<<grid, 256, 0, stream>>>(A, lda, W, bias, act, outF, outB, Mm, Nn, Kk, ldc);
  };

  for (int blk = 0; blk < 2; ++blk) {
    const int o = 5 + 9 * blk;
    const float* conv_w = (const float*)d_in[o + 1];
    const float* conv_b = (const float*)d_in[o + 2];
    const float* dt_b   = (const float*)d_in[o + 5];
    const float* A_log  = (const float*)d_in[o + 6];
    const float* Dp     = (const float*)d_in[o + 7];
    const unsigned short* hin = blk ? h2_bf : h_bf;

    // in_proj: (M,512) x (2048,512)^T -> xz (bf16)
    gemm(hin, 512, wb_in[blk], nullptr, 0, nullptr, xz_bf, M, 2048, 512, 2048);
    // causal depthwise conv + SiLU
    k_conv_silu<<<(M * DI) / 256, 256, 0, stream>>>(xz_bf, conv_w, conv_b, xc_bf, S);
    // x_proj: (M,1024) x (64,1024)^T -> proj (fp32: dt|B|C)
    gemm(xc_bf, 1024, wb_xp[blk], nullptr, 0, proj_f, nullptr, M, 64, 1024, 64);
    // dt slice -> bf16
    k_slice_dt<<<(M * 32 + 255) / 256, 256, 0, stream>>>(proj_f, dtp_bf, M * 32);
    // dt proj + bias + softplus: (M,32) x (1024,32)^T -> dt (bf16)
    gemm(dtp_bf, 32, wb_dt[blk], dt_b, 1, nullptr, dt_bf, M, 1024, 32, 1024);
    // selective scan (+D skip, +SiLU(z) gate) -> gated y (bf16)
    k_scan<<<(B * DI) / 256, 256, 0, stream>>>(dt_bf, xc_bf, proj_f, xz_bf, A_log, Dp, yg_bf, S);
    if (blk == 0) {
      // out_proj for all tokens -> next block's input (bf16)
      gemm(yg_bf, 1024, wb_out[blk], nullptr, 0, nullptr, h2_bf, M, 512, 1024, 512);
    } else {
      // only the last token survives the final LN -> M = B rows, lda = S*1024
      gemm(yg_bf + (size_t)(S - 1) * 1024, (long long)S * 1024, wb_out[blk],
           nullptr, 0, hlast, nullptr, B, 512, 1024, 512);
    }
  }

  k_head<<<B, 512, 0, stream>>>(hlast, norm_g, norm_b, h1_w, h1_b, h2_w, h2_b,
                                x, skip_w, skip_b, (float*)d_out, S);
}
